// PretrainPrev_51453708206797
// MI455X (gfx1250) — matrix operand order repaired and checked by reading, compiled, not run on hardware
//
#include <hip/hip_runtime.h>

typedef _Float16 v16h __attribute__((ext_vector_type(16)));
typedef _Float16 v8h  __attribute__((ext_vector_type(8)));
typedef float    v8f  __attribute__((ext_vector_type(8)));
typedef int      v4i  __attribute__((ext_vector_type(4)));
typedef int      v8i  __attribute__((ext_vector_type(8)));

constexpr int BB   = 8;
constexpr int CIN  = 960;
constexpr int HH   = 64;
constexpr int WW   = 64;
constexpr int NPIX = BB * HH * WW;   // 32768
constexpr int DD   = 512;
constexpr int KCLS = 2;
constexpr int MM   = 10;

constexpr int KC   = 64;    // K-chunk staged via TDM (divides 960 and 512)
constexpr int ROWB = 144;   // padded LDS row stride in bytes (128 data + 16 pad)

__device__ __forceinline__ float wredsum(float v) {
#pragma unroll
  for (int o = 16; o; o >>= 1) v += __shfl_xor(v, o, 32);
  return v;
}

// ---------------------------------------------------------------------------
// Tensor Data Mover: DMA one KCx64 f16 weight tile (rows stride 'ldk' elems)
// into LDS at 'lds_dst', padding each 32-dword row with 4 dwords so the LDS
// row stride is 144B (bank-conflict-free, 16B aligned).  CDNA5 ISA ch. 8.
// ---------------------------------------------------------------------------
__device__ __forceinline__ void tdm_load_panel(void* lds_dst, const _Float16* gsrc,
                                               int ldk) {
  union { unsigned int w[4]; v4i v; } g0;
  union { unsigned int w[8]; v8i v; } g1;
  const unsigned long long ga = (unsigned long long)(uintptr_t)gsrc;
  const unsigned int lo = (unsigned int)(uintptr_t)lds_dst;
  // --- D# group 0 (ISA 8.3) ---
  g0.w[0] = 1u;                                            // count=1, user desc
  g0.w[1] = lo;                                            // lds_addr (bytes)
  g0.w[2] = (unsigned int)ga;                              // global_addr[31:0]
  g0.w[3] = (unsigned int)((ga >> 32) & 0x01FFFFFFu)       // global_addr[56:32]
          | (2u << 30);                                    // type = 2 ("image")
  // --- D# group 1 (ISA 8.4) ---
  // data_size=1 (2B), pad_enable, pad_interval=32 dwords (code 4),
  // pad_amount=4 dwords (code 3), workgroup_mask=0 (not in cluster)
  g1.w[0] = (1u << 16) | (1u << 20) | (4u << 22) | (3u << 25);
  const unsigned int td0 = (unsigned int)KC;               // tensor_dim0 (elems)
  const unsigned int td1 = 64u;                            // tensor_dim1 (rows)
  const unsigned long long st0 = (unsigned long long)ldk;  // dim0 stride (elems)
  g1.w[1] = (td0 << 16);                                   // tensor_dim0[15:0]
  g1.w[2] = (td0 >> 16) | (td1 << 16);                     // dim0[31:16]|dim1[15:0]
  g1.w[3] = (td1 >> 16) | ((unsigned int)KC << 16);        // dim1[31:16]|tile_dim0
  g1.w[4] = 64u;                                           // tile_dim1=64, tile_dim2=0
  g1.w[5] = (unsigned int)st0;                             // stride0[31:0]
  g1.w[6] = (unsigned int)(st0 >> 32) & 0xFFFFu;           // stride0[47:32]
  g1.w[7] = 0u;
  asm volatile("tensor_load_to_lds %0, %1" :: "s"(g0.v), "s"(g1.v) : "memory");
}

// ---------------------------------------------------------------------------
// Weight conversion f32 -> f16 (layouts already [d][k] row-major = OIHW 1x1)
// ---------------------------------------------------------------------------
__global__ void cvt_weights(const float* __restrict__ pw, const float* __restrict__ pj,
                            _Float16* __restrict__ W1, _Float16* __restrict__ W2) {
  int i = blockIdx.x * blockDim.x + threadIdx.x;
  const int n1 = DD * CIN;
  const int n2 = DD * DD;
  if (i < n1)           W1[i]      = (_Float16)pw[i];
  else if (i < n1 + n2) W2[i - n1] = (_Float16)pj[i - n1];
}

// ---------------------------------------------------------------------------
// protos -> l2-normalized protos (20 rows of 512, one wave each)
// ---------------------------------------------------------------------------
__global__ void proto_norm(const float* __restrict__ protos, float* __restrict__ PN) {
  const int r = blockIdx.x;       // 0..19  (k*10+m)
  const int lane = threadIdx.x;   // 32
  const float* src = protos + (size_t)r * DD;
  float ss = 0.f;
#pragma unroll
  for (int j = 0; j < 16; ++j) { float v = src[lane + 32 * j]; ss += v * v; }
  float inv = 1.f / fmaxf(sqrtf(wredsum(ss)), 1e-12f);
#pragma unroll
  for (int j = 0; j < 16; ++j)
    PN[(size_t)r * DD + lane + 32 * j] = src[lane + 32 * j] * inv;
}

// ---------------------------------------------------------------------------
// Depthwise 3x3 SAME conv; emit A matrix as f16 in [pixel][channel] row-major
// ---------------------------------------------------------------------------
__global__ void dw_conv(const float* __restrict__ feats,
                        const float* __restrict__ dww,
                        _Float16* __restrict__ A16) {
  const int c = blockIdx.y * blockDim.x + threadIdx.x;  // 0..959
  const int bh = blockIdx.x;
  const int b = bh >> 6, h = bh & 63;
  float w9[9];
#pragma unroll
  for (int i = 0; i < 9; ++i) w9[i] = dww[c * 9 + i];
  const float* chan = feats + ((size_t)(b * CIN + c)) * HH * WW;
  for (int w = 0; w < WW; ++w) {
    float s = 0.f;
#pragma unroll
    for (int dh = -1; dh <= 1; ++dh) {
      int hh = h + dh;
      if (hh < 0 || hh >= HH) continue;
      const float* rp = chan + hh * WW;
#pragma unroll
      for (int dw = -1; dw <= 1; ++dw) {
        int w2 = w + dw;
        if (w2 < 0 || w2 >= WW) continue;
        s += rp[w2] * w9[(dh + 1) * 3 + (dw + 1)];
      }
    }
    A16[((size_t)(b * HH * WW) + h * WW + w) * CIN + c] = (_Float16)s;
  }
}

// ---------------------------------------------------------------------------
// WMMA GEMM with TDM-staged, double-buffered weight panels in LDS.
// 256 threads = 8 waves; block tile 128(M) x 64(N); wave tile 16x64.
// ---------------------------------------------------------------------------
__global__ void __launch_bounds__(256)
gemm_wmma(const _Float16* __restrict__ A,
          const _Float16* __restrict__ Wt,
          const float* __restrict__ bias,
          const float* __restrict__ scale,
          const float* __restrict__ shift,
          _Float16* __restrict__ outh,
          float* __restrict__ outf,
          int Ka) {
  __shared__ __align__(16) char panel[2][64 * ROWB];   // 2 x 9 KB
  const int lane = threadIdx.x & 31;
  const int wid  = threadIdx.x >> 5;
  const int lm = lane & 15;   // A row / B column within tile
  const int lh = lane >> 4;   // K-half selector
  const int n0 = blockIdx.x * 128 + wid * 16;
  const int d0 = blockIdx.y * 64;

  v8f acc[4];
#pragma unroll
  for (int t = 0; t < 4; ++t)
#pragma unroll
    for (int i = 0; i < 8; ++i) acc[t][i] = 0.f;

  const int nchunks = Ka / KC;
  if (threadIdx.x < 32)                        // wave 0 drives the TDM
    tdm_load_panel(panel[0], Wt + (size_t)d0 * Ka, Ka);

  const _Float16* arow = A + (size_t)(n0 + lm) * Ka + lh * 16;

  for (int c = 0; c < nchunks; ++c) {
    if (threadIdx.x < 32) {
      if (c + 1 < nchunks) {
        tdm_load_panel(panel[(c + 1) & 1],
                       Wt + (size_t)d0 * Ka + (c + 1) * KC, Ka);
        __builtin_amdgcn_s_wait_tensorcnt(1);  // chunk c landed, c+1 in flight
      } else {
        __builtin_amdgcn_s_wait_tensorcnt(0);
      }
    }
    __syncthreads();                           // publish panel[c&1]
    const char* buf = panel[c & 1];
    const int kbase = c * KC;
#pragma unroll
    for (int ks = 0; ks < KC; ks += 32) {
      union { v16h v; v8h h[2]; } af;
      // lanes 0-15: K {0..15}; lanes 16-31: K {16..31}
      af.h[0] = *reinterpret_cast<const v8h*>(arow + kbase + ks);
      af.h[1] = *reinterpret_cast<const v8h*>(arow + kbase + ks + 8);
      // Hoist all four B fragments before the WMMA chain so the backend can
      // clause the 8 ds_load_b128 and overlap them with the matrix pipe.
      v16h bf[4];
#pragma unroll
      for (int t = 0; t < 4; ++t)
        bf[t] = *reinterpret_cast<const v16h*>(
            buf + (t * 16 + lm) * ROWB + ks * 2 + lh * 32);
#pragma unroll
      for (int t = 0; t < 4; ++t)
        acc[t] = __builtin_amdgcn_wmma_f32_16x16x32_f16(
            false, af.v, false, bf[t], (short)0, acc[t], false, false);
    }
    if (c + 1 < nchunks)
      __builtin_prefetch(arow + (c + 1) * KC, 0, 3);   // global_prefetch_b8
    __syncthreads();                           // done reading panel[c&1]
  }

#pragma unroll
  for (int t = 0; t < 4; ++t) {
    const int d = d0 + t * 16 + lm;
    const float bv = bias ? bias[d] : 0.f;
    const float sv = scale ? scale[d] : 1.f;
    const float hv = shift ? shift[d] : 0.f;
#pragma unroll
    for (int v = 0; v < 8; ++v) {
      const int row = v + lh * 8;              // C/D: VGPR v holds M=v / M=v+8
      const float val = (acc[t][v] + bv) * sv + hv;
      const size_t idx = (size_t)(n0 + row) * DD + d;
      if (outh) outh[idx] = (_Float16)val;
      else      outf[idx] = val;
    }
  }
}

// ---------------------------------------------------------------------------
// Per-pixel: LayerNorm(512) -> l2norm -> cosine sims vs 20 protos -> max over M
// -> LayerNorm(K=2) -> out_seg + pred.  One wave per pixel row.
// ---------------------------------------------------------------------------
__global__ void ln_l2_masks(float* __restrict__ Cbuf,
                            const float* __restrict__ ln1s, const float* __restrict__ ln1b,
                            const float* __restrict__ PN,
                            const float* __restrict__ ln2s, const float* __restrict__ ln2b,
                            float* __restrict__ masks, int* __restrict__ pred,
                            float* __restrict__ out_seg) {
  const int lane = threadIdx.x & 31;
  const int wid  = threadIdx.x >> 5;
  const int n = blockIdx.x * 8 + wid;
  float* row = Cbuf + (size_t)n * DD;
  float x[16];
#pragma unroll
  for (int j = 0; j < 16; ++j) x[j] = row[lane + 32 * j];
  float s = 0.f;
#pragma unroll
  for (int j = 0; j < 16; ++j) s += x[j];
  const float mean = wredsum(s) * (1.f / DD);
  float vv = 0.f;
#pragma unroll
  for (int j = 0; j < 16; ++j) { float d = x[j] - mean; vv += d * d; }
  const float rstd = rsqrtf(wredsum(vv) * (1.f / DD) + 1e-5f);
  float ss = 0.f;
#pragma unroll
  for (int j = 0; j < 16; ++j) {
    const int d = lane + 32 * j;
    x[j] = (x[j] - mean) * rstd * ln1s[d] + ln1b[d];
    ss += x[j] * x[j];
  }
  const float inv = 1.f / fmaxf(sqrtf(wredsum(ss)), 1e-12f);
#pragma unroll
  for (int j = 0; j < 16; ++j) { x[j] *= inv; row[lane + 32 * j] = x[j]; }

  float mx[2] = {-1e30f, -1e30f};
  for (int k = 0; k < KCLS; ++k)
    for (int m = 0; m < MM; ++m) {
      const float* pp = PN + ((size_t)(k * MM + m)) * DD;
      float dp = 0.f;
#pragma unroll
      for (int j = 0; j < 16; ++j) dp += x[j] * pp[lane + 32 * j];
      dp = wredsum(dp);
      if (lane == 0) masks[(size_t)n * 20 + k * MM + m] = dp;
      mx[k] = fmaxf(mx[k], dp);
    }
  const float m2 = 0.5f * (mx[0] + mx[1]);
  const float d0 = mx[0] - m2, d1 = mx[1] - m2;
  const float r2 = rsqrtf(0.5f * (d0 * d0 + d1 * d1) + 1e-5f);
  const float s0 = d0 * r2 * ln2s[0] + ln2b[0];
  const float s1 = d1 * r2 * ln2s[1] + ln2b[1];
  if (lane == 0) {
    const int b = n >> 12, hw = n & 4095;
    out_seg[(size_t)b * (KCLS * 4096) + 0 * 4096 + hw] = s0;
    out_seg[(size_t)b * (KCLS * 4096) + 1 * 4096 + hw] = s1;
    pred[n] = (s1 > s0) ? 1 : 0;
  }
}

// ---------------------------------------------------------------------------
// Masked Sinkhorn (multiplicative scale-factor form) + hard argmax assignment.
// ---------------------------------------------------------------------------
__global__ void __launch_bounds__(1024)
sinkhorn(const float* __restrict__ masks, const int* __restrict__ gt,
         const int* __restrict__ pred, float* __restrict__ ews,
         int* __restrict__ mstar, int* __restrict__ cnts) {
  const int k = blockIdx.x;
  const int tid = threadIdx.x;
  const int lane = tid & 31, wid = tid >> 5;
  __shared__ float sdata[32 * MM];
  __shared__ float aS[MM];
  __shared__ float scal[2];
  __shared__ int cntS[MM];
  constexpr int NPT = NPIX / 1024;  // 32
  float bv[NPT];
  float ssum = 0.f, bnv = 0.f;
  float* ek = ews + (size_t)k * NPIX * MM;
  for (int i = 0; i < NPT; ++i) {
    const int n = tid + 1024 * i;
    const float sel = (gt[n] == k) ? 1.f : 0.f;
    bnv += sel;
    const float* mrow = masks + (size_t)n * 20 + k * MM;
#pragma unroll
    for (int m = 0; m < MM; ++m) {
      const float e = sel * __expf(mrow[m] * 20.0f);  // 1/EPS_SK
      ek[(size_t)n * MM + m] = e;
      ssum += e;
    }
    bv[i] = 1.f;
  }
  ssum = wredsum(ssum); bnv = wredsum(bnv);
  if (lane == 0) { sdata[wid] = ssum; sdata[32 + wid] = bnv; }
  __syncthreads();
  if (tid == 0) {
    float a = 0.f, b = 0.f;
    for (int w = 0; w < 32; ++w) { a += sdata[w]; b += sdata[32 + w]; }
    scal[0] = a; scal[1] = fmaxf(b, 1.f);
  }
  __syncthreads();
  if (tid < MM) { aS[tid] = 1.f / fmaxf(scal[0], 1e-12f); cntS[tid] = 0; }
  __syncthreads();
  const float Bn = scal[1];
  for (int it = 0; it < 3; ++it) {
    float rs[MM];
#pragma unroll
    for (int m = 0; m < MM; ++m) rs[m] = 0.f;
    for (int i = 0; i < NPT; ++i) {
      const float* ep = ek + (size_t)(tid + 1024 * i) * MM;
#pragma unroll
      for (int m = 0; m < MM; ++m) rs[m] += ep[m] * bv[i];
    }
#pragma unroll
    for (int m = 0; m < MM; ++m) {
      const float r = wredsum(rs[m]);
      if (lane == 0) sdata[wid * MM + m] = r;
    }
    __syncthreads();
    if (tid < MM) {
      float srow = 0.f;
      for (int w = 0; w < 32; ++w) srow += sdata[w * MM + tid];
      const float R = aS[tid] * srow;
      aS[tid] = aS[tid] / (fmaxf(R, 1e-12f) * (float)MM);
    }
    __syncthreads();
    for (int i = 0; i < NPT; ++i) {
      const float* ep = ek + (size_t)(tid + 1024 * i) * MM;
      float cs = 0.f;
#pragma unroll
      for (int m = 0; m < MM; ++m) cs += ep[m] * aS[m];
      cs *= bv[i];
      bv[i] = ((cs > 0.f) ? bv[i] / fmaxf(cs, 1e-12f) : 0.f) / Bn;
    }
    __syncthreads();
  }
  for (int i = 0; i < NPT; ++i) {
    const int n = tid + 1024 * i;
    const float* ep = ek + (size_t)n * MM;
    int mb = 0; float best = -1.f;
    if (bv[i] > 0.f) {
#pragma unroll
      for (int m = 0; m < MM; ++m) {
        const float q = ep[m] * aS[m];
        if (q > best) { best = q; mb = m; }   // first-max wins (matches argmax)
      }
    }
    const bool mk = (pred[n] == gt[n]) && (gt[n] == k);
    mstar[(size_t)k * NPIX + n] = mk ? mb : -1;
    if (mk) atomicAdd(&cntS[mb], 1);
  }
  __syncthreads();
  if (tid < MM) cnts[k * MM + tid] = cntS[tid];
}

// ---------------------------------------------------------------------------
// f[k][m][d] = sum over pixels assigned to (k,m) of _c[n][d]
// ---------------------------------------------------------------------------
__global__ void f_accum(const float* __restrict__ Cbuf,
                        const int* __restrict__ mstar, float* __restrict__ F) {
  const int k = blockIdx.x / MM;
  const int m = blockIdx.x % MM;
  const int d = threadIdx.x;  // 512
  const int* ms = mstar + (size_t)k * NPIX;
  float acc = 0.f;
  for (int n = 0; n < NPIX; ++n)
    if (ms[n] == m) acc += Cbuf[(size_t)n * DD + d];
  F[(size_t)blockIdx.x * DD + d] = acc;
}

__device__ __forceinline__ float blkredsum256(float v, float* red) {
  v = wredsum(v);
  const int lane = threadIdx.x & 31, wid = threadIdx.x >> 5;
  if (lane == 0) red[wid] = v;
  __syncthreads();
  float s = 0.f;
  for (int w = 0; w < 8; ++w) s += red[w];
  __syncthreads();
  return s;
}

// ---------------------------------------------------------------------------
// EMA prototype update + double l2norm -> d_out
// ---------------------------------------------------------------------------
__global__ void proto_final(const float* __restrict__ F, const int* __restrict__ cnts,
                            const float* __restrict__ PN, float* __restrict__ outp) {
  __shared__ float red[8];
  const int r = blockIdx.x;   // k*10+m
  const int k = r / MM;
  const int t = threadIdx.x;  // 256, 2 elems each
  const float f0 = F[(size_t)r * DD + t];
  const float f1 = F[(size_t)r * DD + t + 256];
  const float ss = blkredsum256(f0 * f0 + f1 * f1, red);
  const float inv = 1.f / fmaxf(sqrtf(ss), 1e-12f);
  int nsum = 0;
  for (int m = 0; m < MM; ++m) nsum += cnts[k * MM + m];
  const bool doUpd = (cnts[r] > 0) && (nsum > 0);
  const float p0 = PN[(size_t)r * DD + t];
  const float p1 = PN[(size_t)r * DD + t + 256];
  const float u0 = doUpd ? (0.999f * p0 + 0.001f * f0 * inv) : p0;
  const float u1 = doUpd ? (0.999f * p1 + 0.001f * f1 * inv) : p1;
  const float ss2 = blkredsum256(u0 * u0 + u1 * u1, red);
  const float inv2 = 1.f / fmaxf(sqrtf(ss2), 1e-12f);
  outp[(size_t)r * DD + t]       = u0 * inv2;
  outp[(size_t)r * DD + t + 256] = u1 * inv2;
}

// ---------------------------------------------------------------------------
// Fused UpBlock: 2x2 s2 transposed conv (computed on the fly) + 4 branches
// ---------------------------------------------------------------------------
__global__ void upblock(const float* __restrict__ seg,
                        const float* __restrict__ tw, const float* __restrict__ tb,
                        const float* __restrict__ w1, const float* __restrict__ b1,
                        const float* __restrict__ w2, const float* __restrict__ b2,
                        float* __restrict__ out) {
  const int idx = blockIdx.x * blockDim.x + threadIdx.x;  // 131072
  const int b = idx >> 14;
  const int rem = idx & 16383;
  const int y = rem >> 7, x = rem & 127;
  float u[2][3][3];
#pragma unroll
  for (int dy = 0; dy < 3; ++dy)
#pragma unroll
    for (int dx = 0; dx < 3; ++dx) {
      const int yy = y + dy - 1, xx = x + dx - 1;
      if (yy < 0 || yy >= 128 || xx < 0 || xx >= 128) {
        u[0][dy][dx] = 0.f; u[1][dy][dx] = 0.f;
        continue;
      }
      const int h = yy >> 1, w = xx >> 1;
      const int i = 1 - (yy & 1), j = 1 - (xx & 1);
      const float s0 = seg[(size_t)b * 8192 + 0 * 4096 + h * 64 + w];
      const float s1 = seg[(size_t)b * 8192 + 1 * 4096 + h * 64 + w];
#pragma unroll
      for (int o = 0; o < 2; ++o)
        u[o][dy][dx] = tb[o] + s0 * tw[(0 * 2 + o) * 4 + i * 2 + j]
                             + s1 * tw[(1 * 2 + o) * 4 + i * 2 + j];
    }
#pragma unroll
  for (int br = 0; br < 4; ++br) {
    float t[2];
#pragma unroll
    for (int o = 0; o < 2; ++o) {
      float acc = b1[br * 2 + o];
#pragma unroll
      for (int c = 0; c < 2; ++c)
#pragma unroll
        for (int dy = 0; dy < 3; ++dy)
#pragma unroll
          for (int dx = 0; dx < 3; ++dx)
            acc += u[c][dy][dx] * w1[(((br * 2 + o) * 2 + c) * 3 + dy) * 3 + dx];
      t[o] = fmaxf(acc, 0.f);
    }
#pragma unroll
    for (int o = 0; o < 2; ++o) {
      const float r = b2[br * 2 + o] + t[0] * w2[(br * 2 + o) * 2 + 0]
                                    + t[1] * w2[(br * 2 + o) * 2 + 1];
      out[(size_t)br * 262144 + ((size_t)(b * 2 + o) * 128 + y) * 128 + x] = r;
    }
  }
}

// ---------------------------------------------------------------------------
extern "C" void kernel_launch(void* const* d_in, const int* in_sizes, int n_in,
                              void* d_out, int out_size, void* d_ws, size_t ws_size,
                              hipStream_t stream) {
  (void)in_sizes; (void)n_in; (void)out_size; (void)ws_size;
  const float* feats  = (const float*)d_in[0];
  const int*   gt     = (const int*)d_in[1];
  const float* dw_w   = (const float*)d_in[2];
  const float* pw_w   = (const float*)d_in[3];
  const float* pw_b   = (const float*)d_in[4];
  const float* bn_s   = (const float*)d_in[5];
  const float* bn_b   = (const float*)d_in[6];
  const float* proj_w = (const float*)d_in[7];
  const float* proj_b = (const float*)d_in[8];
  const float* ln1s   = (const float*)d_in[9];
  const float* ln1b   = (const float*)d_in[10];
  const float* protos = (const float*)d_in[11];
  const float* ln2s   = (const float*)d_in[12];
  const float* ln2b   = (const float*)d_in[13];
  const float* tw     = (const float*)d_in[14];
  const float* tb     = (const float*)d_in[15];
  const float* bw1    = (const float*)d_in[16];
  const float* bb1    = (const float*)d_in[17];
  const float* bw2    = (const float*)d_in[18];
  const float* bb2    = (const float*)d_in[19];

  char* ws = (char*)d_ws;
  size_t off = 0;
  auto alloc = [&](size_t bytes) {
    void* p = ws + off;
    off = (off + bytes + 255) & ~(size_t)255;
    return p;
  };
  _Float16* A16 = (_Float16*)alloc((size_t)NPIX * CIN * 2);   // 63 MB
  _Float16* W1  = (_Float16*)alloc((size_t)DD * CIN * 2);
  _Float16* X16 = (_Float16*)alloc((size_t)NPIX * DD * 2);    // 34 MB
  _Float16* W2  = (_Float16*)alloc((size_t)DD * DD * 2);
  float* Cbuf   = (float*)alloc((size_t)NPIX * DD * 4);       // 67 MB (c, then _c in place)
  float* PN     = (float*)alloc((size_t)KCLS * MM * DD * 4);
  float* MASKS  = (float*)alloc((size_t)NPIX * 20 * 4);
  int*   PRED   = (int*)alloc((size_t)NPIX * 4);
  float* EWS    = (float*)alloc((size_t)KCLS * NPIX * MM * 4);
  int*   MSTAR  = (int*)alloc((size_t)KCLS * NPIX * 4);
  float* FBUF   = (float*)alloc((size_t)KCLS * MM * DD * 4);
  int*   CNTS   = (int*)alloc(256);

  float* out     = (float*)d_out;
  float* out_seg = out + 4 * 262144;   // 5th output
  float* out_pr  = out_seg + 65536;    // 6th output

  cvt_weights<<<(DD * CIN + DD * DD) / 256, 256, 0, stream>>>(pw_w, proj_w, W1, W2);
  proto_norm<<<KCLS * MM, 32, 0, stream>>>(protos, PN);
  dw_conv<<<dim3(BB * HH, 3), 320, 0, stream>>>(feats, dw_w, A16);
  // pointwise 960->512 with pw bias + folded BN, f16 out
  gemm_wmma<<<dim3(NPIX / 128, DD / 64), 256, 0, stream>>>(
      A16, W1, pw_b, bn_s, bn_b, X16, nullptr, CIN);
  // proj 512->512 with bias, f32 out
  gemm_wmma<<<dim3(NPIX / 128, DD / 64), 256, 0, stream>>>(
      X16, W2, proj_b, nullptr, nullptr, nullptr, Cbuf, DD);
  ln_l2_masks<<<NPIX / 8, 256, 0, stream>>>(Cbuf, ln1s, ln1b, PN, ln2s, ln2b,
                                            MASKS, PRED, out_seg);
  sinkhorn<<<KCLS, 1024, 0, stream>>>(MASKS, gt, PRED, EWS, MSTAR, CNTS);
  f_accum<<<KCLS * MM, DD, 0, stream>>>(Cbuf, MSTAR, FBUF);
  proto_final<<<KCLS * MM, 256, 0, stream>>>(FBUF, CNTS, PN, out_pr);
  upblock<<<(BB * 128 * 128) / 256, 256, 0, stream>>>(out_seg, tw, tb, bw1, bb1,
                                                      bw2, bb2, out);
}
